// GRU_996432412983
// MI455X (gfx1250) — compile-verified, matrix operand order checked by reading, not run on hardware
//
#include <hip/hip_runtime.h>
#include <hip/hip_bf16.h>
#include <math.h>

#define VOCAB   32000
#define DIM     1024
#define D2      2048
#define NLAYERS 4
#define BB      8
#define LL      512

typedef __attribute__((ext_vector_type(16))) __bf16 v16bf;
typedef __attribute__((ext_vector_type(8)))  __bf16 v8bf;
typedef __attribute__((ext_vector_type(8)))  float  v8f;

// Build a 16x16x32 bf16 A/B fragment for one lane.
// ISA layout (16-bit A 16x32): lanes 0-15 hold K = [0..7] and [16..23],
// lanes 16-31 hold K = [8..15] and [24..31]  ->  two 16-byte chunks at
// koff = (lane>>4)*8 and koff+16 from the row pointer.
__device__ __forceinline__ v16bf load_frag(const __bf16* __restrict__ row, int koff) {
    v8bf lo = *(const v8bf*)(row + koff);
    v8bf hi = *(const v8bf*)(row + koff + 16);
    v16bf f;
#pragma unroll
    for (int i = 0; i < 8; ++i) { f[i] = lo[i]; f[i + 8] = hi[i]; }
    return f;
}

__device__ __forceinline__ v8f wmma_bf16(v16bf a, v16bf b, v8f c) {
    return __builtin_amdgcn_wmma_f32_16x16x32_bf16(false, a, false, b, (short)0, c,
                                                   false, false);
}

// CDNA5 async global->LDS copy (ASYNCcnt-tracked, no VGPR staging).
// lds_off: byte offset within the workgroup's LDS allocation.
__device__ __forceinline__ void async_copy_b128(unsigned lds_off, const void* gsrc) {
    asm volatile("global_load_async_to_lds_b128 %0, %1, off"
                 :: "v"(lds_off), "v"(gsrc) : "memory");
}
__device__ __forceinline__ void wait_asynccnt0() {
    asm volatile("s_wait_asynccnt 0x0" ::: "memory");
}

// ---------------------------------------------------------------- conversions
__global__ void f32_to_bf16_kernel(const float* __restrict__ src,
                                   __bf16* __restrict__ dst, long long n) {
    long long i = (long long)blockIdx.x * blockDim.x + threadIdx.x;
    long long stride = (long long)gridDim.x * blockDim.x;
    for (; i < n; i += stride) dst[i] = (__bf16)src[i];
}

__global__ void embed_gather_kernel(const int* __restrict__ ids,
                                    const float* __restrict__ embed_W,
                                    __bf16* __restrict__ xg) {
    long long i = (long long)blockIdx.x * blockDim.x + threadIdx.x;  // over B*L*D
    long long row = i >> 10;          // b*L + l
    int d = (int)(i & 1023);
    int id = ids[row];
    xg[i] = (__bf16)embed_W[(long long)id * DIM + d];
}

// ------------------------------------------------------- recurrent GRU kernel
// 4 workgroups, one per layer, pipelined over timesteps via global flags.
// Block = 1024 threads = 32 waves. Batch rows 0..7 live, rows 8..15 zero pad.
__launch_bounds__(1024, 1)
__global__ void gru_pipeline_kernel(const __bf16* __restrict__ xg,      // [B,L,D]
                                    const __bf16* __restrict__ gateWb,  // [Ly,2D,2D]
                                    const __bf16* __restrict__ candWb,  // [Ly,D,2D]
                                    const float*  __restrict__ gate_b,  // [Ly,2D]
                                    const float*  __restrict__ cand_b,  // [Ly,D]
                                    const float*  __restrict__ h_start, // [Ly,D]
                                    __bf16* __restrict__ streams,       // [3,L,8,D]
                                    __bf16* __restrict__ ys,            // rows b*L+t
                                    int* __restrict__ flags) {          // [4,L]
    extern __shared__ __align__(128) char smem[];
    __bf16* Abuf  = (__bf16*)smem;                                  // [16][2048] bf16
    float*  hbuf  = (float*)(smem + 16 * D2 * 2);                   // [16][1024] f32
    __bf16* zbuf  = (__bf16*)(smem + 16 * D2 * 2 + 16 * DIM * 4);   // [16][1024] bf16
    __bf16* rhbuf = zbuf + 16 * DIM;                                // [16][1024] bf16

    const int layer = blockIdx.x;
    const int tid   = threadIdx.x;
    const int lane  = tid & 31;
    const int wave  = tid >> 5;            // 0..31
    const int l15   = lane & 15;
    const int koff  = (lane >> 4) * 8;
    const int mbase = (lane >> 4) * 8;

    // zero pad rows 8..15 of A (x and h regions) and init h from h_starter
    for (int i = tid; i < 8 * D2; i += 1024) {
        int m = 8 + (i / D2), k = i % D2;
        Abuf[m * D2 + k] = (__bf16)0.f;
    }
    for (int d = tid; d < DIM; d += 1024) {
        float hs = h_start[layer * DIM + d];
#pragma unroll
        for (int m = 0; m < 8; ++m) {
            hbuf[m * DIM + d] = hs;
            Abuf[m * D2 + DIM + d] = (__bf16)hs;
        }
#pragma unroll
        for (int m = 8; m < 16; ++m) hbuf[m * DIM + d] = 0.f;
    }
    __syncthreads();

    const __bf16* Arow  = Abuf + l15 * D2;
    const __bf16* RHrow = rhbuf + l15 * DIM;

    // each thread async-copies one 16B chunk of x_t (8*DIM bf16 = 1024 chunks)
    const int cidx = tid * 8;                 // bf16 element index into [8][1024]
    const int cm   = cidx >> 10;
    const int cd   = cidx & 1023;
    const unsigned clds = (unsigned)((cm * D2 + cd) * 2);  // Abuf at smem offset 0

    for (int t = 0; t < LL; ++t) {
        // -------- acquire x_t into Abuf rows 0..7, cols 0..1023 (async to LDS)
        if (layer > 0) {
            if (tid == 0) {
                while (__hip_atomic_load(&flags[(layer - 1) * LL + t],
                                         __ATOMIC_ACQUIRE,
                                         __HIP_MEMORY_SCOPE_AGENT) == 0)
                    __builtin_amdgcn_s_sleep(1);
            }
            __syncthreads();
            const __bf16* src = streams + ((size_t)(layer - 1) * LL + t) * (8 * DIM);
            async_copy_b128(clds, src + cidx);
        } else {
            async_copy_b128(clds, xg + ((size_t)cm * LL + t) * DIM + cd);
        }
        wait_asynccnt0();
        __syncthreads();

        // -------- gate GEMM: g = [x|h] @ gate_W^T ; 4 n-tiles per wave, joint
        {
            const __bf16* Wp[4];
#pragma unroll
            for (int tt = 0; tt < 4; ++tt)
                Wp[tt] = gateWb + ((size_t)layer * D2 + wave * 64 + tt * 16 + l15) * D2;
            v8f acc[4] = {};
            v16bf a = load_frag(Arow, koff);
            v16bf b[4];
#pragma unroll
            for (int tt = 0; tt < 4; ++tt) b[tt] = load_frag(Wp[tt], koff);
            for (int k0 = 32; k0 < D2; k0 += 32) {          // software pipelined
                v16bf an = load_frag(Arow + k0, koff);
                v16bf bn[4];
#pragma unroll
                for (int tt = 0; tt < 4; ++tt) bn[tt] = load_frag(Wp[tt] + k0, koff);
#pragma unroll
                for (int tt = 0; tt < 4; ++tt) acc[tt] = wmma_bf16(a, b[tt], acc[tt]);
                a = an;
#pragma unroll
                for (int tt = 0; tt < 4; ++tt) b[tt] = bn[tt];
            }
#pragma unroll
            for (int tt = 0; tt < 4; ++tt) acc[tt] = wmma_bf16(a, b[tt], acc[tt]);

#pragma unroll
            for (int tt = 0; tt < 4; ++tt) {
                int n = wave * 64 + tt * 16 + l15;
                float bias = gate_b[layer * D2 + n];
#pragma unroll
                for (int j = 0; j < 8; ++j) {
                    int m = mbase + j;
                    float s = 1.f / (1.f + __expf(-(acc[tt][j] + bias)));
                    if (n < DIM) {
                        zbuf[m * DIM + n] = (__bf16)s;                 // z
                    } else {
                        int nn = n - DIM;                              // r * h
                        rhbuf[m * DIM + nn] = (__bf16)(s * hbuf[m * DIM + nn]);
                    }
                }
            }
        }
        __syncthreads();

        // -------- cand GEMM: tanh([x|r*h] @ cand_W^T) ; 2 n-tiles per wave
        {
            const __bf16* Wp[2];
#pragma unroll
            for (int tt = 0; tt < 2; ++tt)
                Wp[tt] = candWb + ((size_t)layer * DIM + wave * 32 + tt * 16 + l15) * D2;
            v8f acc[2] = {};
            v16bf a = load_frag(Arow, koff);
            v16bf b[2];
#pragma unroll
            for (int tt = 0; tt < 2; ++tt) b[tt] = load_frag(Wp[tt], koff);
            for (int k0 = 32; k0 < D2; k0 += 32) {          // software pipelined
                const __bf16* ap = (k0 < DIM) ? (Arow + k0) : (RHrow + (k0 - DIM));
                v16bf an = load_frag(ap, koff);
                v16bf bn[2];
#pragma unroll
                for (int tt = 0; tt < 2; ++tt) bn[tt] = load_frag(Wp[tt] + k0, koff);
#pragma unroll
                for (int tt = 0; tt < 2; ++tt) acc[tt] = wmma_bf16(a, b[tt], acc[tt]);
                a = an;
#pragma unroll
                for (int tt = 0; tt < 2; ++tt) b[tt] = bn[tt];
            }
#pragma unroll
            for (int tt = 0; tt < 2; ++tt) acc[tt] = wmma_bf16(a, b[tt], acc[tt]);

#pragma unroll
            for (int tt = 0; tt < 2; ++tt) {
                int n = wave * 32 + tt * 16 + l15;
                float bias = cand_b[layer * DIM + n];
#pragma unroll
                for (int j = 0; j < 8; ++j) {
                    int m = mbase + j;
                    float c = tanhf(acc[tt][j] + bias);
                    float z = (float)zbuf[m * DIM + n];
                    float hn = z * hbuf[m * DIM + n] + (1.f - z) * c;
                    if (m < 8) {                                    // keep pad rows zero
                        hbuf[m * DIM + n] = hn;
                        Abuf[m * D2 + DIM + n] = (__bf16)hn;        // h for t+1
                        if (layer < 3)
                            streams[((size_t)layer * LL + t) * (8 * DIM) + m * DIM + n] =
                                (__bf16)hn;
                        else
                            ys[((size_t)m * LL + t) * DIM + n] = (__bf16)hn;
                    }
                }
            }
        }
        __syncthreads();
        if (layer < 3 && tid == 0)
            __hip_atomic_store(&flags[layer * LL + t], 1, __ATOMIC_RELEASE,
                               __HIP_MEMORY_SCOPE_AGENT);
    }
}

// -------------------------------------------------------------- logits GEMM
// C[4096,32000] = ys[4096,1024] @ out_W^T.  Block 256 thr = 8 waves,
// wave tile 32x64 (2 m-tiles x 4 n-tiles), workgroup tile 64x256.
__launch_bounds__(256)
__global__ void logits_gemm_kernel(const __bf16* __restrict__ ys,
                                   const __bf16* __restrict__ outWb,
                                   float* __restrict__ out) {
    const int tid  = threadIdx.x;
    const int lane = tid & 31;
    const int wave = tid >> 5;           // 0..7
    const int wm   = wave >> 2;          // 0..1
    const int wn   = wave & 3;           // 0..3
    const int l15  = lane & 15;
    const int koff = (lane >> 4) * 8;
    const int mbase = (lane >> 4) * 8;

    const int m0 = blockIdx.y * 64 + wm * 32;
    const int n0 = blockIdx.x * 256 + wn * 64;

    v8f acc[2][4] = {};
    const __bf16* Ap[2];
    const __bf16* Bp[4];
    Ap[0] = ys + (size_t)(m0 + l15) * DIM;
    Ap[1] = ys + (size_t)(m0 + 16 + l15) * DIM;
#pragma unroll
    for (int i = 0; i < 4; ++i) Bp[i] = outWb + (size_t)(n0 + i * 16 + l15) * DIM;

    v16bf a[2], b[4];
#pragma unroll
    for (int i = 0; i < 2; ++i) a[i] = load_frag(Ap[i], koff);
#pragma unroll
    for (int i = 0; i < 4; ++i) b[i] = load_frag(Bp[i], koff);

    for (int k0 = 32; k0 < DIM; k0 += 32) {          // software pipelined
        __builtin_prefetch((const void*)(Bp[0] + k0 + 64), 0, 1);  // global_prefetch_b8
        v16bf an[2], bn[4];
#pragma unroll
        for (int i = 0; i < 2; ++i) an[i] = load_frag(Ap[i] + k0, koff);
#pragma unroll
        for (int i = 0; i < 4; ++i) bn[i] = load_frag(Bp[i] + k0, koff);
#pragma unroll
        for (int im = 0; im < 2; ++im)
#pragma unroll
            for (int in_ = 0; in_ < 4; ++in_)
                acc[im][in_] = wmma_bf16(a[im], b[in_], acc[im][in_]);
#pragma unroll
        for (int i = 0; i < 2; ++i) a[i] = an[i];
#pragma unroll
        for (int i = 0; i < 4; ++i) b[i] = bn[i];
    }
#pragma unroll
    for (int im = 0; im < 2; ++im)
#pragma unroll
        for (int in_ = 0; in_ < 4; ++in_)
            acc[im][in_] = wmma_bf16(a[im], b[in_], acc[im][in_]);

#pragma unroll
    for (int im = 0; im < 2; ++im)
#pragma unroll
        for (int in_ = 0; in_ < 4; ++in_)
#pragma unroll
            for (int j = 0; j < 8; ++j) {
                int m = m0 + im * 16 + mbase + j;
                int n = n0 + in_ * 16 + l15;
                out[(size_t)m * VOCAB + n] = acc[im][in_][j];
            }
}

// ------------------------------------------------------------------ launcher
extern "C" void kernel_launch(void* const* d_in, const int* in_sizes, int n_in,
                              void* d_out, int out_size, void* d_ws, size_t ws_size,
                              hipStream_t stream) {
    const int*   ids       = (const int*)d_in[0];
    const float* embed_W   = (const float*)d_in[1];
    const float* h_starter = (const float*)d_in[2];
    const float* gate_W    = (const float*)d_in[3];
    const float* gate_b    = (const float*)d_in[4];
    const float* cand_W    = (const float*)d_in[5];
    const float* cand_b    = (const float*)d_in[6];
    const float* out_W     = (const float*)d_in[7];
    float*       logits    = (float*)d_out;

    char*  ws  = (char*)d_ws;
    size_t off = 0;
    auto alloc = [&](size_t bytes) {
        char* p = ws + off;
        off += (bytes + 255) & ~(size_t)255;
        return p;
    };
    __bf16* xg      = (__bf16*)alloc((size_t)BB * LL * DIM * 2);
    __bf16* gateWb  = (__bf16*)alloc((size_t)NLAYERS * D2 * D2 * 2);
    __bf16* candWb  = (__bf16*)alloc((size_t)NLAYERS * DIM * D2 * 2);
    __bf16* outWb   = (__bf16*)alloc((size_t)VOCAB * DIM * 2);
    __bf16* streams = (__bf16*)alloc((size_t)3 * LL * 8 * DIM * 2);
    __bf16* ysb     = (__bf16*)alloc((size_t)BB * LL * DIM * 2);
    int*    flags   = (int*)alloc((size_t)NLAYERS * LL * 4);

    hipMemsetAsync(flags, 0, (size_t)NLAYERS * LL * 4, stream);

    f32_to_bf16_kernel<<<4096, 256, 0, stream>>>(gate_W, gateWb,
                                                 (long long)NLAYERS * D2 * D2);
    f32_to_bf16_kernel<<<4096, 256, 0, stream>>>(cand_W, candWb,
                                                 (long long)NLAYERS * DIM * D2);
    f32_to_bf16_kernel<<<4096, 256, 0, stream>>>(out_W, outWb,
                                                 (long long)VOCAB * DIM);
    embed_gather_kernel<<<(BB * LL * DIM) / 256, 256, 0, stream>>>(ids, embed_W, xg);

    size_t lds = (size_t)16 * D2 * 2      // A  [16][2048] bf16
               + (size_t)16 * DIM * 4     // h  [16][1024] f32
               + (size_t)16 * DIM * 2     // z  [16][1024] bf16
               + (size_t)16 * DIM * 2;    // rh [16][1024] bf16  -> 192 KB (<320 KB WGP)
    gru_pipeline_kernel<<<dim3(NLAYERS), dim3(1024), lds, stream>>>(
        xg, gateWb, candWb, gate_b, cand_b, h_starter, streams, ysb, flags);

    logits_gemm_kernel<<<dim3(VOCAB / 256, (BB * LL) / 64), dim3(256), 0, stream>>>(
        ysb, outWb, logits);
}
